// SimCLR_V2_74079595921585
// MI455X (gfx1250) — matrix operand order repaired and checked by reading, not compiled, not run on hardware
//
#include <hip/hip_runtime.h>
#include <hip/hip_bf16.h>
#include <stdint.h>

// ---------------------------------------------------------------------------
// SimCLR forward on MI455X (gfx1250): bf16 WMMA GEMMs + fixed-shift logsumexp.
// ---------------------------------------------------------------------------

typedef __attribute__((ext_vector_type(16))) __bf16 v16bf;
typedef __attribute__((ext_vector_type(8)))  float  v8f;

#define B_ROWS 16384
#define D_DIM  2048
#define H_DIM  2048
#define F_DIM  128
#define INV_TEMP (1.0f / 0.07f)     // also the fixed logsumexp shift C (sim/T <= 1/T)
#define LOG2E    1.4426950408889634f
#define NEG_INF_F (-1e30f)

__device__ __forceinline__ unsigned short f32_to_bf16_rte(float f) {
    unsigned int u = __float_as_uint(f);
    u += 0x7FFFu + ((u >> 16) & 1u);   // round-to-nearest-even
    return (unsigned short)(u >> 16);
}

union FragU {            // one bf16 WMMA operand fragment: 16 x bf16 = 8 VGPRs
    v16bf v;
    uint4 q[2];
};

// --------------------------- conversion kernels ----------------------------

__global__ void cvt_bf16_kernel(const float* __restrict__ src,
                                unsigned short* __restrict__ dst, int n) {
    for (int i = blockIdx.x * blockDim.x + threadIdx.x; i < n;
         i += gridDim.x * blockDim.x)
        dst[i] = f32_to_bf16_rte(src[i]);
}

// src [rows, cols] fp32 -> dst [cols, rows] bf16 (transpose so every WMMA
// B-fragment is two contiguous 16B loads).
__global__ void cvt_bf16_T_kernel(const float* __restrict__ src,
                                  unsigned short* __restrict__ dst,
                                  int rows, int cols) {
    int n = rows * cols;
    for (int i = blockIdx.x * blockDim.x + threadIdx.x; i < n;
         i += gridDim.x * blockDim.x) {
        int k    = i % rows;   // source row
        int ncol = i / rows;   // source col == dest row
        dst[i] = f32_to_bf16_rte(src[(size_t)k * cols + ncol]);
    }
}

// ------------------------------ WMMA GEMM ----------------------------------
// C[M,N] = act(A[M,K] @ BT[N,K]^T + bias), one wave computes a 32x64 strip:
// 2 A-fragments x 4 B-fragments -> 8 WMMAs per k-step from 12 b128 loads
// (1.5 loads/WMMA). All fragments of a k-step are loaded before the WMMA
// chain so loads clause and overlap the matrix pipe.
// MODE 0: bf16 output with ReLU (h).   MODE 1: fp32 output, no ReLU (p).

template <int MODE>
__global__ void __launch_bounds__(256)
wmma_gemm_kernel(const unsigned short* __restrict__ A,
                 const unsigned short* __restrict__ BT,
                 const float* __restrict__ bias,
                 unsigned short* __restrict__ Cbf,
                 float* __restrict__ Cf,
                 int M, int N, int K) {
    const int lane    = threadIdx.x & 31;
    const int gw      = blockIdx.x * (blockDim.x >> 5) + (threadIdx.x >> 5);
    const int nGrp    = N >> 6;                 // 64-wide column groups
    const int rowTile = gw / nGrp;              // 32-row tiles
    const int colGrp  = gw % nGrp;
    if (rowTile >= (M >> 5)) return;

    const int halfSel = lane >> 4;              // 0: lanes 0-15, 1: lanes 16-31
    const int l16     = lane & 15;
    const int aK0     = halfSel ? 8 : 0;        // A frag K-chunk base per half
    const int bKb     = halfSel ? 8 : 0;        // B frag K base per half
    const int colBase = colGrp * 64;

    v8f acc[2][4];
#pragma unroll
    for (int r = 0; r < 2; ++r)
#pragma unroll
        for (int t = 0; t < 4; ++t) acc[r][t] = (v8f){0, 0, 0, 0, 0, 0, 0, 0};

    const unsigned short* aRowPtr0 = A + (size_t)(rowTile * 32 + l16) * K;
    const unsigned short* aRowPtr1 = aRowPtr0 + (size_t)16 * K;
    const unsigned short* bRowPtr[4];
#pragma unroll
    for (int t = 0; t < 4; ++t)
        bRowPtr[t] = BT + (size_t)(colBase + t * 16 + l16) * K;

    for (int k = 0; k < K; k += 32) {
        FragU a0, a1, b[4];
        a0.q[0] = *reinterpret_cast<const uint4*>(aRowPtr0 + k + aK0);
        a0.q[1] = *reinterpret_cast<const uint4*>(aRowPtr0 + k + aK0 + 16);
        a1.q[0] = *reinterpret_cast<const uint4*>(aRowPtr1 + k + aK0);
        a1.q[1] = *reinterpret_cast<const uint4*>(aRowPtr1 + k + aK0 + 16);
#pragma unroll
        for (int t = 0; t < 4; ++t) {
            b[t].q[0] = *reinterpret_cast<const uint4*>(bRowPtr[t] + k + bKb);
            b[t].q[1] = *reinterpret_cast<const uint4*>(bRowPtr[t] + k + bKb + 16);
        }
#pragma unroll
        for (int t = 0; t < 4; ++t) {
            acc[0][t] = __builtin_amdgcn_wmma_f32_16x16x32_bf16(
                false, a0.v, false, b[t].v, (short)0, acc[0][t], false, false);
            acc[1][t] = __builtin_amdgcn_wmma_f32_16x16x32_bf16(
                false, a1.v, false, b[t].v, (short)0, acc[1][t], false, false);
        }
    }

#pragma unroll
    for (int r = 0; r < 2; ++r) {
        const int rowOut = rowTile * 32 + r * 16 + (halfSel ? 8 : 0);
#pragma unroll
        for (int t = 0; t < 4; ++t) {
            int col  = colBase + t * 16 + l16;
            float bv = bias[col];
#pragma unroll
            for (int v = 0; v < 8; ++v) {
                float x = acc[r][t][v] + bv;
                int row = rowOut + v;
                if (MODE == 0) {
                    x = fmaxf(x, 0.0f);
                    Cbf[(size_t)row * N + col] = f32_to_bf16_rte(x);
                } else {
                    Cf[(size_t)row * N + col] = x;
                }
            }
        }
    }
}

// ------------------------- row L2 normalization ----------------------------
// One wave per row of p (128 floats). Writes fp32 z to d_out (+1 offset is
// only 4B-aligned, so scalar stores) and packed bf16 z for the sim GEMM.

__global__ void normalize_kernel(const float* __restrict__ P,
                                 float* __restrict__ Zout,
                                 unsigned short* __restrict__ Zb) {
    int wave = (blockIdx.x * blockDim.x + threadIdx.x) >> 5;
    int lane = threadIdx.x & 31;
    if (wave >= B_ROWS) return;
    float4 x = reinterpret_cast<const float4*>(P + (size_t)wave * F_DIM)[lane];
    float ss = x.x * x.x + x.y * x.y + x.z * x.z + x.w * x.w;
#pragma unroll
    for (int off = 1; off < 32; off <<= 1) ss += __shfl_xor(ss, off, 32);
    float inv = 1.0f / fmaxf(sqrtf(ss), 1e-12f);
    float zx = x.x * inv, zy = x.y * inv, zz = x.z * inv, zw = x.w * inv;
    float* zr = Zout + (size_t)wave * F_DIM + lane * 4;
    zr[0] = zx; zr[1] = zy; zr[2] = zz; zr[3] = zw;
    uint2 packed;
    packed.x = (unsigned)f32_to_bf16_rte(zx) | ((unsigned)f32_to_bf16_rte(zy) << 16);
    packed.y = (unsigned)f32_to_bf16_rte(zz) | ((unsigned)f32_to_bf16_rte(zw) << 16);
    reinterpret_cast<uint2*>(Zb + (size_t)wave * F_DIM)[lane] = packed;
}

// -------------------- rolled positives: pos_i = z_i . z_{i+N} --------------
// One wave per row; fp32 z read from d_out (scalar loads: base is 4B-aligned).

__global__ void pos_kernel(const float* __restrict__ Z,
                           float* __restrict__ rowPos) {
    int row  = (blockIdx.x * blockDim.x + threadIdx.x) >> 5;
    int lane = threadIdx.x & 31;
    if (row >= B_ROWS) return;
    int partner = (row + (B_ROWS / 2)) & (B_ROWS - 1);
    const float* a = Z + (size_t)row * F_DIM + lane * 4;
    const float* b = Z + (size_t)partner * F_DIM + lane * 4;
    float d = a[0] * b[0] + a[1] * b[1] + a[2] * b[2] + a[3] * b[3];
#pragma unroll
    for (int off = 1; off < 32; off <<= 1) d += __shfl_xor(d, off, 32);
    if (lane == 0) rowPos[row] = d * INV_TEMP;
}

// ------------- sim = z z^T / T, fixed-shift masked logsumexp ---------------
// Block owns 16 rows; 8 waves sweep the 1024 column tiles, two tiles per
// iteration (two independent WMMA accumulator chains sharing the A frags).
// Rows of z are unit vectors => sim/T <= 1/T =: C, so
//   lse_i = C + log( sum_{j != i} exp(sim_ij/T - C) )
// needs no online max tracking. exp is evaluated as v_exp_f32 (2^x) with the
// whole affine argument folded into one v_fmaak:
//   exp(sim/T - C) = exp2( sim * (log2e/T) - C*log2e ).
// Per element: fmaak + cndmask(diag -> -1e30, exp2 -> 0) + exp2 + add.

__global__ void __launch_bounds__(256)
sim_lse_kernel(const unsigned short* __restrict__ Zb,
               float* __restrict__ rowLse) {
    __shared__ float smS[8][16];

    const int lane    = threadIdx.x & 31;
    const int wave    = threadIdx.x >> 5;
    const int l16     = lane & 15;
    const int halfSel = lane >> 4;
    const int rowTile = blockIdx.x;                 // 0..1023
    const int aK0     = halfSel ? 8 : 0;
    const int bKb     = halfSel ? 8 : 0;
    const int rowBase = rowTile * 16;
    const int myRowLo = halfSel ? 8 : 0;

    const float S2 = INV_TEMP * LOG2E;              // scale into exp2 domain
    // (shift is also C*log2e == S2 since C == INV_TEMP)

    // A fragments (this block's 16 rows of z), loaded once: K=128 -> 4 steps.
    FragU afr[4];
    const unsigned short* aRowPtr = Zb + (size_t)(rowBase + l16) * F_DIM;
#pragma unroll
    for (int s = 0; s < 4; ++s) {
        afr[s].q[0] = *reinterpret_cast<const uint4*>(aRowPtr + s * 32 + aK0);
        afr[s].q[1] = *reinterpret_cast<const uint4*>(aRowPtr + s * 32 + aK0 + 16);
    }

    float ssum[8];
#pragma unroll
    for (int v = 0; v < 8; ++v) ssum[v] = 0.0f;

    for (int ct = wave * 2; ct < (B_ROWS / 16); ct += 16) {
        const unsigned short* b0p = Zb + (size_t)(ct * 16 + l16) * F_DIM;
        const unsigned short* b1p = b0p + 16 * F_DIM;
        v8f acc0 = (v8f){0, 0, 0, 0, 0, 0, 0, 0};
        v8f acc1 = (v8f){0, 0, 0, 0, 0, 0, 0, 0};
#pragma unroll
        for (int s = 0; s < 4; ++s) {
            FragU b0, b1;
            b0.q[0] = *reinterpret_cast<const uint4*>(b0p + s * 32 + bKb);
            b0.q[1] = *reinterpret_cast<const uint4*>(b0p + s * 32 + bKb + 16);
            b1.q[0] = *reinterpret_cast<const uint4*>(b1p + s * 32 + bKb);
            b1.q[1] = *reinterpret_cast<const uint4*>(b1p + s * 32 + bKb + 16);
            acc0 = __builtin_amdgcn_wmma_f32_16x16x32_bf16(
                false, afr[s].v, false, b0.v, (short)0, acc0, false, false);
            acc1 = __builtin_amdgcn_wmma_f32_16x16x32_bf16(
                false, afr[s].v, false, b1.v, (short)0, acc1, false, false);
        }
        const int col0 = ct * 16 + l16;
        const int col1 = col0 + 16;
#pragma unroll
        for (int v = 0; v < 8; ++v) {
            const int grow = rowBase + myRowLo + v;
            float x0 = __builtin_fmaf(acc0[v], S2, -S2);    // exp2 argument
            float x1 = __builtin_fmaf(acc1[v], S2, -S2);
            x0 = (col0 != grow) ? x0 : NEG_INF_F;           // cndmask, exp2->0
            x1 = (col1 != grow) ? x1 : NEG_INF_F;
            ssum[v] += __builtin_amdgcn_exp2f(x0) + __builtin_amdgcn_exp2f(x1);
        }
    }

    // Sum the 16 lanes of each half (same row, disjoint column subsets).
#pragma unroll
    for (int v = 0; v < 8; ++v) {
#pragma unroll
        for (int off = 1; off < 16; off <<= 1)
            ssum[v] += __shfl_xor(ssum[v], off, 32);
    }
    if (l16 == 0) {
#pragma unroll
        for (int v = 0; v < 8; ++v) smS[wave][myRowLo + v] = ssum[v];
    }
    __syncthreads();
    if (threadIdx.x < 16) {                          // sum 8 wave partials
        int r = threadIdx.x;
        float t = 0.0f;
#pragma unroll
        for (int w = 0; w < 8; ++w) t += smS[w][r];
        rowLse[rowBase + r] = __logf(t) + INV_TEMP;  // undo fixed shift
    }
}

// ------------------------------- final loss --------------------------------

__global__ void loss_kernel(const float* __restrict__ rowLse,
                            const float* __restrict__ rowPos,
                            float* __restrict__ out) {
    __shared__ float red[256];
    float s = 0.0f;
    for (int i = threadIdx.x; i < B_ROWS; i += 256)
        s += rowLse[i] - rowPos[i];
    red[threadIdx.x] = s;
    __syncthreads();
    for (int off = 128; off > 0; off >>= 1) {
        if (threadIdx.x < off) red[threadIdx.x] += red[threadIdx.x + off];
        __syncthreads();
    }
    if (threadIdx.x == 0) out[0] = red[0] / (float)B_ROWS;
}

// ------------------------------- launcher ----------------------------------

extern "C" void kernel_launch(void* const* d_in, const int* in_sizes, int n_in,
                              void* d_out, int out_size, void* d_ws,
                              size_t ws_size, hipStream_t stream) {
    const float* features = (const float*)d_in[0];  // [16384, 2048]
    const float* w1       = (const float*)d_in[1];  // [2048, 2048]
    const float* b1       = (const float*)d_in[2];  // [2048]
    const float* w2       = (const float*)d_in[3];  // [2048, 128]
    const float* b2       = (const float*)d_in[4];  // [128]
    float* out = (float*)d_out;                     // [1 + 16384*128]

    // Workspace layout (all offsets 16B-aligned), ~149 MB total.
    char* ws = (char*)d_ws;
    unsigned short* Xb  = (unsigned short*)(ws + 0);          // 16384x2048 bf16
    unsigned short* W1T = (unsigned short*)(ws + 67108864);   //  2048x2048 bf16 (H x D)
    unsigned short* W2T = (unsigned short*)(ws + 75497472);   //   128x2048 bf16 (F x H)
    unsigned short* Hb  = (unsigned short*)(ws + 76021760);   // 16384x2048 bf16
    float*          P   = (float*)         (ws + 143130624);  // 16384x128  f32
    unsigned short* Zb  = (unsigned short*)(ws + 151519232);  // 16384x128  bf16
    float*       rowLse = (float*)         (ws + 155713536);  // 16384 f32
    float*       rowPos = (float*)         (ws + 155779072);  // 16384 f32

    // 1) fp32 -> bf16 (weights transposed for contiguous B-fragments).
    cvt_bf16_kernel  <<<4096, 256, 0, stream>>>(features, Xb, B_ROWS * D_DIM);
    cvt_bf16_T_kernel<<<2048, 256, 0, stream>>>(w1, W1T, D_DIM, H_DIM);
    cvt_bf16_T_kernel<<< 256, 256, 0, stream>>>(w2, W2T, H_DIM, F_DIM);

    // 2) h = relu(X @ W1 + b1): (16384/32)*(2048/64) = 16384 waves -> 2048 blocks.
    wmma_gemm_kernel<0><<<2048, 256, 0, stream>>>(Xb, W1T, b1, Hb, nullptr,
                                                  B_ROWS, H_DIM, D_DIM);
    // 3) p = h @ W2 + b2: (16384/32)*(128/64) = 1024 waves -> 128 blocks.
    wmma_gemm_kernel<1><<<128, 256, 0, stream>>>(Hb, W2T, b2, nullptr, P,
                                                 B_ROWS, F_DIM, H_DIM);
    // 4) z = p / ||p||: fp32 rows into d_out[1..], bf16 copy for sim GEMM.
    normalize_kernel<<<2048, 256, 0, stream>>>(P, out + 1, Zb);
    // 5) pos_i = z_i . z_{i+N} / T (branchless sim kernel doesn't capture it).
    pos_kernel<<<2048, 256, 0, stream>>>(out + 1, rowPos);
    // 6) masked fixed-shift row logsumexp of z z^T / T.
    sim_lse_kernel<<<B_ROWS / 16, 256, 0, stream>>>(Zb, rowLse);
    // 7) loss = mean(lse - pos).
    loss_kernel<<<1, 256, 0, stream>>>(rowLse, rowPos, out);
}
